// rpn_regression_84980222919088
// MI455X (gfx1250) — matrix operand order repaired and checked by reading, not compiled, not run on hardware
//
#include <hip/hip_runtime.h>

typedef __attribute__((ext_vector_type(16))) _Float16 v16h;
typedef __attribute__((ext_vector_type(8)))  float    v8f;
typedef __attribute__((ext_vector_type(4)))  int      v4i;

// ---------------- problem dims ----------------
constexpr int BN = 8, CN = 512, HN = 64, WN = 64;
constexpr int P  = HN * WN;                                // 4096
// output layout (flat f32): feat | cls_prob | bbox | zero | zero
constexpr long long FEAT_N   = (long long)BN * CN * P;     // 16777216
constexpr long long CLS_OFF  = FEAT_N;
constexpr long long BOX_OFF  = CLS_OFF + (long long)BN * 18 * P;   // 17367040
constexpr long long ZERO_OFF = BOX_OFF + (long long)BN * 36 * P;   // 18546688
// workspace layout (halves)
constexpr long long X16_OFF = 0;                                   // base_feat f16 [B][512][P]
constexpr long long R16_OFF = X16_OFF + (long long)BN * 512 * P;   // convf_rpn f16 [B][128][P]
constexpr long long C16_OFF = R16_OFF + (long long)BN * 128 * P;   // rpn_conv1 f16 [B][384][P]
constexpr long long WG1_OFF = C16_OFF + (long long)BN * 384 * P;   // 16*32 frags
constexpr long long WG2_OFF = WG1_OFF + 512LL * 512;               // 9*4*24 frags
constexpr long long WG3_OFF = WG2_OFF + 864LL * 512;               // 12*4 frags

// ---------------- CDNA5 async global->LDS staging ----------------
#if __has_builtin(__builtin_amdgcn_global_load_async_to_lds_b128)
#define HAS_ASYNC_LDS 1
#else
#define HAS_ASYNC_LDS 0
#warning "gfx1250: __builtin_amdgcn_global_load_async_to_lds_b128 not available; using synchronous staging fallback"
#endif

typedef __attribute__((address_space(1))) v4i* as1_v4i_p;
typedef __attribute__((address_space(3))) v4i* as3_v4i_p;

__device__ inline void g2l_b128(_Float16* l, const _Float16* g) {
#if HAS_ASYNC_LDS
  __builtin_amdgcn_global_load_async_to_lds_b128((as1_v4i_p)g, (as3_v4i_p)l, 0, 0);
#else
  *reinterpret_cast<uint4*>(l) = *reinterpret_cast<const uint4*>(g);
#endif
}

__device__ inline void wait_async_all() {
#if HAS_ASYNC_LDS
#if __has_builtin(__builtin_amdgcn_s_wait_asynccnt)
  __builtin_amdgcn_s_wait_asynccnt(0);
#else
  asm volatile("s_wait_asynccnt 0" ::: "memory");
#endif
#endif
}

// ---------------- WMMA fragment helpers (ISA 7.12.2, wave32) ----------------
// A fragment straight from fragment-ready global weights: [frag][lane][16] halves.
__device__ inline v16h load_a_frag(const _Float16* __restrict__ wg, int fragIdx, int lane) {
  return *reinterpret_cast<const v16h*>(wg + ((long long)fragIdx * 32 + lane) * 16);
}
// B 32x16: lane%16 = N column, 16 K halves in table order.
__device__ inline v16h lds_load_b(const _Float16* Bt, int lane, int nOff, int stride) {
  const _Float16* p = Bt + nOff + (lane & 15);
  v16h b;
#pragma unroll
  for (int i = 0; i < 16; ++i) b[i] = p[((i & 7) + 16 * (i >> 3) + 8 * (lane >> 4)) * stride];
  return b;
}

#define WMMA_F16(A, B, C) \
  __builtin_amdgcn_wmma_f32_16x16x32_f16(false, (A), false, (B), (short)0, (C), false, false)

// =====================================================================
// Kernel 0: one-shot f32->f16 conversion of base_feat + fragment-ready
// repack of all conv weights (matches load_a_frag layout).
// =====================================================================
__global__ __launch_bounds__(256)
void prep_kernel(const float* __restrict__ base,
                 const float* __restrict__ wf, const float* __restrict__ w2,
                 const float* __restrict__ wc1,
                 const float* __restrict__ wcls, const float* __restrict__ wbox,
                 _Float16* __restrict__ x16, _Float16* __restrict__ wg1,
                 _Float16* __restrict__ wg2, _Float16* __restrict__ wg3) {
  const long long tid = (long long)blockIdx.x * blockDim.x + threadIdx.x;
  const long long stride = (long long)gridDim.x * blockDim.x;
  // base_feat: 8,388,608 float4 -> f16
  for (long long i = tid; i < 8388608LL; i += stride) {
    float4 v = reinterpret_cast<const float4*>(base)[i];
    _Float16* d = x16 + i * 4;
    d[0] = (_Float16)v.x; d[1] = (_Float16)v.y; d[2] = (_Float16)v.z; d[3] = (_Float16)v.w;
  }
  // wg1: concat(w_f,w_2) [512x512] -> frags (kc<16, mt<32)
  for (long long i = tid; i < 262144LL; i += stride) {
    int e = i & 15, lane = (i >> 4) & 31; int rest = (int)(i >> 9);
    int mt = rest & 31, kc = rest >> 5;
    int m = mt * 16 + (lane & 15);
    int K = (e & 7) + 16 * (e >> 3) + 8 * (lane >> 4);
    int c = kc * 32 + K;
    float v = (m < 128) ? wf[(long long)m * 512 + c] : w2[(long long)(m - 128) * 512 + c];
    wg1[i] = (_Float16)v;
  }
  // wg2: w_c1 [384][128][9] -> frags ((tap*4+kc)*24 + mt)
  for (long long i = tid; i < 442368LL; i += stride) {
    int e = i & 15, lane = (i >> 4) & 31; int rest = (int)(i >> 9);
    int mt = rest % 24, q = rest / 24;      // q = tap*4 + kc
    int kc = q & 3, tap = q >> 2;
    int m = mt * 16 + (lane & 15);
    int K = (e & 7) + 16 * (e >> 3) + 8 * (lane >> 4);
    int c = kc * 32 + K;
    wg2[i] = (_Float16)wc1[((long long)m * 128 + c) * 9 + tap];
  }
  // wg3: [w_cls;w_box;pad] [64x384] -> frags (kc<12, mt<4)
  for (long long i = tid; i < 24576LL; i += stride) {
    int e = i & 15, lane = (i >> 4) & 31; int rest = (int)(i >> 9);
    int mt = rest & 3, kc = rest >> 2;
    int m = mt * 16 + (lane & 15);
    int K = (e & 7) + 16 * (e >> 3) + 8 * (lane >> 4);
    int c = kc * 32 + K;
    float v = 0.f;
    if (m < 18)      v = wcls[(long long)m * 384 + c];
    else if (m < 54) v = wbox[(long long)(m - 18) * 384 + c];
    wg3[i] = (_Float16)v;
  }
}

// =====================================================================
// Kernel 1: feat = relu([w_f;w_2] @ base_feat + b). M=512,K=512,N=4096/b.
// Block 64M x 128N, 4 waves (2x2), wave = 32M x 64N = 8 wmma / K-step.
// B staged async double-buffered; A frags direct from repacked global.
// =====================================================================
__global__ __launch_bounds__(128)
void feat_gemm_kernel(const _Float16* __restrict__ x16, const _Float16* __restrict__ wg1,
                      const float* __restrict__ bf, const float* __restrict__ b2,
                      float* __restrict__ feat, _Float16* __restrict__ r16) {
  __shared__ __align__(16) _Float16 Bt[2][32 * 128];
  const int t = threadIdx.x, lane = t & 31, wv = t >> 5;
  const int pBase = blockIdx.x * 128;
  const int mBase = blockIdx.y * 64;
  const int b     = blockIdx.z;
  const int waveM = (wv >> 1) * 32, waveN = (wv & 1) * 64;
  const int mt0   = (mBase + waveM) >> 4;

  v8f acc[2][4] = {};

  auto stage = [&](int buf, int kcIdx) {
    const _Float16* src = x16 + ((long long)(b * 512 + kcIdx * 32)) * P + pBase;
    for (int i = t; i < 512; i += 128) {
      int k = i >> 4, n8 = (i & 15) * 8;
      g2l_b128(&Bt[buf][k * 128 + n8], src + (long long)k * P + n8);
    }
  };

  stage(0, 0);
  for (int kcIdx = 0; kcIdx < 16; ++kcIdx) {
    wait_async_all();
    __syncthreads();
    if (kcIdx + 1 < 16) stage((kcIdx + 1) & 1, kcIdx + 1);
    const _Float16* Bc = Bt[kcIdx & 1];
    v16h a0 = load_a_frag(wg1, kcIdx * 32 + mt0, lane);
    v16h a1 = load_a_frag(wg1, kcIdx * 32 + mt0 + 1, lane);
#pragma unroll
    for (int j = 0; j < 4; ++j) {
      v16h bj = lds_load_b(Bc, lane, waveN + 16 * j, 128);
      acc[0][j] = WMMA_F16(a0, bj, acc[0][j]);
      acc[1][j] = WMMA_F16(a1, bj, acc[1][j]);
    }
  }

  const int hi = lane >> 4, lo = lane & 15;
#pragma unroll
  for (int mi = 0; mi < 2; ++mi)
#pragma unroll
    for (int nj = 0; nj < 4; ++nj)
#pragma unroll
      for (int r = 0; r < 8; ++r) {
        int gm = mBase + waveM + 16 * mi + r + 8 * hi;
        int n  = pBase + waveN + 16 * nj + lo;
        float bias = (gm < 128) ? bf[gm] : b2[gm - 128];
        float v = acc[mi][nj][r] + bias;
        v = v > 0.f ? v : 0.f;
        feat[((long long)b * 512 + gm) * P + n] = v;
        if (gm < 128) r16[((long long)b * 128 + gm) * P + n] = (_Float16)v;
      }
}

// =====================================================================
// Kernel 2: 3x3 conv (pad 1), implicit GEMM M=384, K=128*9, N = one row.
// Entire 3-row x 128-ch input strip loaded once (async) into LDS with
// zero-padded borders; tap loop reads B frags directly (shift folded in).
// =====================================================================
__global__ __launch_bounds__(128)
void rpnconv3_kernel(const _Float16* __restrict__ r16, const _Float16* __restrict__ wg2,
                     const float* __restrict__ bc1, _Float16* __restrict__ c16) {
  __shared__ __align__(16) _Float16 SB[128 * 3 * 80];   // 60 KB strip
  const int t = threadIdx.x, lane = t & 31, wv = t >> 5;
  const int row   = blockIdx.x;
  const int mBase = blockIdx.y * 64;
  const int b     = blockIdx.z;
  const int waveM = (wv >> 1) * 32, waveN = (wv & 1) * 32;
  const int mt0   = (mBase + waveM) >> 4;

  // zero strip (borders + out-of-image rows stay zero)
  uint4 zz = make_uint4(0u, 0u, 0u, 0u);
  for (int i = t; i < 3840; i += 128) reinterpret_cast<uint4*>(SB)[i] = zz;
  __syncthreads();
  // async-load interior: pixel w lives at column w+8 (16B-aligned starts)
#pragma unroll
  for (int r = 0; r < 3; ++r) {
    int h = row - 1 + r;
    if ((unsigned)h < 64u) {
      for (int i = t; i < 1024; i += 128) {
        int ch = i >> 3, n8 = (i & 7) * 8;
        g2l_b128(SB + (ch * 3 + r) * 80 + 8 + n8,
                 r16 + ((long long)(b * 128 + ch)) * P + h * 64 + n8);
      }
    }
  }
  wait_async_all();
  __syncthreads();

  v8f acc[2][2] = {};
  for (int tap = 0; tap < 9; ++tap) {
    const int dy = tap / 3 - 1, dx = tap % 3 - 1;
    const int rowIdx = dy + 1;
#pragma unroll
    for (int kc = 0; kc < 4; ++kc) {
      int fragBase = (tap * 4 + kc) * 24 + mt0;
      v16h a0 = load_a_frag(wg2, fragBase, lane);
      v16h a1 = load_a_frag(wg2, fragBase + 1, lane);
      const _Float16* sp = SB + rowIdx * 80 + 8 + dx + (lane & 15);
      v16h b0, b1;
#pragma unroll
      for (int i = 0; i < 16; ++i) b0[i] = sp[(kc * 32 + (i & 7) + 16 * (i >> 3) + 8 * (lane >> 4)) * 240 + waveN];
#pragma unroll
      for (int i = 0; i < 16; ++i) b1[i] = sp[(kc * 32 + (i & 7) + 16 * (i >> 3) + 8 * (lane >> 4)) * 240 + waveN + 16];
      acc[0][0] = WMMA_F16(a0, b0, acc[0][0]);
      acc[0][1] = WMMA_F16(a0, b1, acc[0][1]);
      acc[1][0] = WMMA_F16(a1, b0, acc[1][0]);
      acc[1][1] = WMMA_F16(a1, b1, acc[1][1]);
    }
  }

  const int hi = lane >> 4, lo = lane & 15;
#pragma unroll
  for (int mi = 0; mi < 2; ++mi)
#pragma unroll
    for (int nj = 0; nj < 2; ++nj)
#pragma unroll
      for (int r = 0; r < 8; ++r) {
        int gm = mBase + waveM + 16 * mi + r + 8 * hi;
        int n  = waveN + 16 * nj + lo;
        float v = acc[mi][nj][r] + bc1[gm];
        v = v > 0.f ? v : 0.f;
        c16[((long long)b * 384 + gm) * P + row * 64 + n] = (_Float16)v;
      }
}

// =====================================================================
// Kernel 3: fused cls(18)+box(36) 1x1 convs (M padded to 64), K=384,
// N tile 64; in-block paired softmax (c, c+9); zero losses.
// =====================================================================
__global__ __launch_bounds__(128)
void clsbox_kernel(const _Float16* __restrict__ c16, const _Float16* __restrict__ wg3,
                   const float* __restrict__ bcls, const float* __restrict__ bbox,
                   float* __restrict__ out) {
  __shared__ __align__(16) _Float16 Bt[2][32 * 64];
  __shared__ float S[18 * 64];
  const int t = threadIdx.x, lane = t & 31, wv = t >> 5;
  const int pBase = blockIdx.x * 64;
  const int b     = blockIdx.y;
  const int waveM = (wv >> 1) * 32, waveN = (wv & 1) * 32;
  const int mt0   = waveM >> 4;

  v8f acc[2][2] = {};

  auto stage = [&](int buf, int kcIdx) {
    const _Float16* src = c16 + ((long long)(b * 384 + kcIdx * 32)) * P + pBase;
    for (int i = t; i < 256; i += 128) {
      int k = i >> 3, n8 = (i & 7) * 8;
      g2l_b128(&Bt[buf][k * 64 + n8], src + (long long)k * P + n8);
    }
  };

  stage(0, 0);
  for (int kcIdx = 0; kcIdx < 12; ++kcIdx) {
    wait_async_all();
    __syncthreads();
    if (kcIdx + 1 < 12) stage((kcIdx + 1) & 1, kcIdx + 1);
    const _Float16* Bc = Bt[kcIdx & 1];
    v16h a0 = load_a_frag(wg3, kcIdx * 4 + mt0, lane);
    v16h a1 = load_a_frag(wg3, kcIdx * 4 + mt0 + 1, lane);
    v16h b0 = lds_load_b(Bc, lane, waveN, 64);
    v16h b1 = lds_load_b(Bc, lane, waveN + 16, 64);
    acc[0][0] = WMMA_F16(a0, b0, acc[0][0]);
    acc[0][1] = WMMA_F16(a0, b1, acc[0][1]);
    acc[1][0] = WMMA_F16(a1, b0, acc[1][0]);
    acc[1][1] = WMMA_F16(a1, b1, acc[1][1]);
  }

  const int hi = lane >> 4, lo = lane & 15;
  float* outCls = out + CLS_OFF;
  float* outBox = out + BOX_OFF;
#pragma unroll
  for (int mi = 0; mi < 2; ++mi)
#pragma unroll
    for (int nj = 0; nj < 2; ++nj)
#pragma unroll
      for (int r = 0; r < 8; ++r) {
        int gm = waveM + 16 * mi + r + 8 * hi;
        int nl = waveN + 16 * nj + lo;
        float v = acc[mi][nj][r];
        if (gm < 18) {
          S[gm * 64 + nl] = v + bcls[gm];
        } else if (gm < 54) {
          outBox[((long long)b * 36 + (gm - 18)) * P + pBase + nl] = v + bbox[gm - 18];
        }
      }
  __syncthreads();
  for (int i = t; i < 576; i += 128) {
    int c = i >> 6, n = i & 63;
    float s0 = S[c * 64 + n], s1 = S[(c + 9) * 64 + n];
    float mx = fmaxf(s0, s1);
    float e0 = __expf(s0 - mx), e1 = __expf(s1 - mx);
    float inv = 1.f / (e0 + e1);
    outCls[((long long)b * 18 + c) * P + pBase + n]       = e0 * inv;
    outCls[((long long)b * 18 + (c + 9)) * P + pBase + n] = e1 * inv;
  }
  if (blockIdx.x == 0 && b == 0 && t == 0) {
    out[ZERO_OFF] = 0.f;
    out[ZERO_OFF + 1] = 0.f;
  }
}

extern "C" void kernel_launch(void* const* d_in, const int* in_sizes, int n_in,
                              void* d_out, int out_size, void* d_ws, size_t ws_size,
                              hipStream_t stream) {
  (void)in_sizes; (void)n_in; (void)out_size; (void)ws_size;
  const float* base = (const float*)d_in[0];
  const float* wf   = (const float*)d_in[3];
  const float* bf   = (const float*)d_in[4];
  const float* w2   = (const float*)d_in[5];
  const float* b2   = (const float*)d_in[6];
  const float* wc1  = (const float*)d_in[7];
  const float* bc1  = (const float*)d_in[8];
  const float* wcls = (const float*)d_in[9];
  const float* bcls = (const float*)d_in[10];
  const float* wbox = (const float*)d_in[11];
  const float* bbox = (const float*)d_in[12];
  float* out = (float*)d_out;

  _Float16* ws  = (_Float16*)d_ws;
  _Float16* x16 = ws + X16_OFF;
  _Float16* r16 = ws + R16_OFF;
  _Float16* c16 = ws + C16_OFF;
  _Float16* wg1 = ws + WG1_OFF;
  _Float16* wg2 = ws + WG2_OFF;
  _Float16* wg3 = ws + WG3_OFF;

  prep_kernel<<<dim3(4096), 256, 0, stream>>>(base, wf, w2, wc1, wcls, wbox,
                                              x16, wg1, wg2, wg3);
  feat_gemm_kernel<<<dim3(32, 8, 8), 128, 0, stream>>>(x16, wg1, bf, b2, out, r16);
  rpnconv3_kernel<<<dim3(64, 6, 8), 128, 0, stream>>>(r16, wg2, bc1, c16);
  clsbox_kernel<<<dim3(64, 8), 128, 0, stream>>>(c16, wg3, bcls, bbox, out);
}
